// TVGTformer_83150566851149
// MI455X (gfx1250) — matrix operand order repaired and checked by reading, not compiled, not run on hardware
//
#include <hip/hip_runtime.h>
#include <hip/hip_bf16.h>
#include <math.h>

#define BB 16
#define TT 1024
#define FF 64
#define HH 8
#define DD 64
#define DFFN 256

typedef __attribute__((ext_vector_type(16))) __bf16 bf16x16;
typedef __attribute__((ext_vector_type(8)))  float  v8f;

#define ZERO8 {0.f,0.f,0.f,0.f,0.f,0.f,0.f,0.f}

union ABf16 {
    bf16x16 v;
    unsigned short u[16];
    uint4 q[2];
};

static __device__ __forceinline__ unsigned short f2bf(float f) {
    unsigned int x = __float_as_uint(f);
    unsigned int r = x + 0x7FFFu + ((x >> 16) & 1u);
    return (unsigned short)(r >> 16);
}

// A-operand: 16x32 bf16 tile, row-major source, row (M) = lane&15.
// ISA A layout: lane<16 holds K {0..7,16..23}, lane>=16 holds K {8..15,24..31}.
static __device__ __forceinline__ ABf16 load_tile16(const unsigned short* base, int ld) {
    int lane = threadIdx.x & 31;
    int r = lane & 15, hi = lane >> 4;
    const unsigned short* p = base + (size_t)r * ld + hi * 8;
    ABf16 a;
    a.q[0] = *(const uint4*)(p);
    a.q[1] = *(const uint4*)(p + 16);
    return a;
}

// B-operand: 32x16 bf16 tile; source stored row-major as B^T (row = N, K contiguous).
// ISA B layout: lane<16 (N=lane) holds K {0..7,16..23}, lane>=16 holds K {8..15,24..31}.
static __device__ __forceinline__ ABf16 load_tileB16(const unsigned short* base, int ld) {
    int lane = threadIdx.x & 31;
    int n = lane & 15, hi = lane >> 4;
    const unsigned short* p = base + (size_t)n * ld + hi * 8;
    ABf16 a;
    a.q[0] = *(const uint4*)(p);
    a.q[1] = *(const uint4*)(p + 16);
    return a;
}

static __device__ __forceinline__ v8f wmma_bf16(ABf16 a, ABf16 b, v8f c) {
    return __builtin_amdgcn_wmma_f32_16x16x32_bf16(false, a.v, false, b.v,
                                                   (short)0, c, false, false);
}

static __device__ __forceinline__ float wred_max(float v) {
#pragma unroll
    for (int m = 16; m >= 1; m >>= 1) v = fmaxf(v, __shfl_xor(v, m, 32));
    return v;
}
static __device__ __forceinline__ float wred_sum(float v) {
#pragma unroll
    for (int m = 16; m >= 1; m >>= 1) v += __shfl_xor(v, m, 32);
    return v;
}

// ---------------- conversion kernels ----------------

__global__ void k_cvt(const float* __restrict__ src, unsigned short* __restrict__ dst, int n) {
    int i = blockIdx.x * 256 + threadIdx.x;
    if (i < n) dst[i] = f2bf(src[i]);
}

// src: [nmat][rows][cols] f32 -> dst: [nmat][cols][rows] bf16 (transpose per matrix)
__global__ void k_cvt_t(const float* __restrict__ src, unsigned short* __restrict__ dst,
                        int rows, int cols) {
    size_t mo = (size_t)blockIdx.y * rows * cols;
    int i = blockIdx.x * 256 + threadIdx.x;
    if (i < rows * cols) {
        int r = i / cols, c = i % cols;
        dst[mo + (size_t)c * rows + r] = f2bf(src[mo + i]);
    }
}

// ---------------- A_soft = softmax(relu(A_temp)) rowwise ----------------

__global__ void k_asoft(const float* __restrict__ At, float* __restrict__ As) {
    __shared__ float red[8];
    int row = blockIdx.x;                      // b*T + t
    const float* src = At + (size_t)row * TT;
    float* dst = As + (size_t)row * TT;
    int tid = threadIdx.x;                     // 256 threads
    int lane = tid & 31, wv = tid >> 5;
    float v[4];
    float mx = -3.0e38f;
#pragma unroll
    for (int j = 0; j < 4; ++j) {
        v[j] = fmaxf(src[tid + j * 256], 0.0f);
        mx = fmaxf(mx, v[j]);
    }
    mx = wred_max(mx);
    if (lane == 0) red[wv] = mx;
    __syncthreads();
    float m = red[0];
#pragma unroll
    for (int i = 1; i < 8; ++i) m = fmaxf(m, red[i]);
    __syncthreads();
    float s = 0.0f;
#pragma unroll
    for (int j = 0; j < 4; ++j) s += __expf(v[j] - m);
    s = wred_sum(s);
    if (lane == 0) red[wv] = s;
    __syncthreads();
    float tot = 0.0f;
#pragma unroll
    for (int i = 0; i < 8; ++i) tot += red[i];
    float inv = 1.0f / tot;
#pragma unroll
    for (int j = 0; j < 4; ++j) dst[tid + j * 256] = __expf(v[j] - m) * inv;
}

// ---------------- QKV projection ----------------
// grid (B*T/16, H), block 96 (wave0=Q, wave1=K, wave2=V)
__global__ void k_qkv(const unsigned short* __restrict__ Xbf,
                      const unsigned short* __restrict__ Wqt,
                      const unsigned short* __restrict__ Wkt,
                      const unsigned short* __restrict__ Wvt,
                      const float* __restrict__ bQ, const float* __restrict__ bK,
                      const float* __restrict__ bV,
                      unsigned short* __restrict__ Qbf,
                      unsigned short* __restrict__ Kbf,
                      unsigned short* __restrict__ Vt) {
    int mt = blockIdx.x, h = blockIdx.y;
    int wv = __builtin_amdgcn_readfirstlane(threadIdx.x >> 5);
    int lane = threadIdx.x & 31;
    int n = lane & 15, hiL = lane >> 4;
    int tg0 = mt * 16;
    int b = tg0 >> 10, t0 = tg0 & 1023;
    const unsigned short* Wt = (wv == 0) ? Wqt : (wv == 1) ? Wkt : Wvt;
    const float* bias = (wv == 0) ? bQ : (wv == 1) ? bK : bV;
    ABf16 a0 = load_tile16(Xbf + (size_t)tg0 * FF, FF);
    ABf16 a1 = load_tile16(Xbf + (size_t)tg0 * FF + 32, FF);
    for (int dt = 0; dt < 4; ++dt) {
        ABf16 w0 = load_tileB16(Wt + (size_t)h * 4096 + (size_t)(dt * 16) * 64, 64);
        ABf16 w1 = load_tileB16(Wt + (size_t)h * 4096 + (size_t)(dt * 16) * 64 + 32, 64);
        v8f acc = ZERO8;
        acc = wmma_bf16(a0, w0, acc);
        acc = wmma_bf16(a1, w1, acc);
        float bia = bias[h * 64 + dt * 16 + n];
        if (wv < 2) {
            unsigned short* dst = (wv == 0 ? Qbf : Kbf) +
                (((size_t)h * BB + b) * TT + t0) * DD + dt * 16;
#pragma unroll
            for (int rr = 0; rr < 8; ++rr) {
                int m = rr + 8 * hiL;
                dst[(size_t)m * DD + n] = f2bf(acc[rr] + bia);
            }
        } else {
            // V stored transposed: Vt[h][b][d][t]
            unsigned short* dst = Vt + (((size_t)h * BB + b) * DD + dt * 16 + n) * TT + t0;
#pragma unroll
            for (int rr = 0; rr < 8; ++rr) dst[rr + 8 * hiL] = f2bf(acc[rr] + bia);
        }
    }
}

// ---------------- attention ----------------
// grid (T/16, H*B), block 128 (4 waves)
__global__ void k_attn(const unsigned short* __restrict__ Qbf,
                       const unsigned short* __restrict__ Kbf,
                       const unsigned short* __restrict__ Vt,
                       const float* __restrict__ As,
                       unsigned short* __restrict__ cat) {
    __shared__ __align__(16) float sc[16][TT];            // 64 KB f32 scores
    __shared__ __align__(16) unsigned short sp[16][TT];   // 32 KB bf16 probs
    int tt = blockIdx.x;
    int hb = blockIdx.y;
    int h = hb >> 4, b = hb & 15;
    int wv = __builtin_amdgcn_readfirstlane(threadIdx.x >> 5);
    int lane = threadIdx.x & 31;
    int n = lane & 15, hiL = lane >> 4;
    int t0 = tt * 16;

    size_t qoff = (((size_t)h * BB + b) * TT + t0) * DD;
    ABf16 aq0 = load_tile16(Qbf + qoff, DD);
    ABf16 aq1 = load_tile16(Qbf + qoff + 32, DD);

    // Phase 1: S = Q K^T / 8, s-tiles split across waves
    for (int st = wv; st < 64; st += 4) {
        int s0 = st * 16;
        size_t koff = (((size_t)h * BB + b) * TT + s0) * DD;
        ABf16 bk0 = load_tileB16(Kbf + koff, DD);
        ABf16 bk1 = load_tileB16(Kbf + koff + 32, DD);
        v8f c = ZERO8;
        c = wmma_bf16(aq0, bk0, c);
        c = wmma_bf16(aq1, bk1, c);
#pragma unroll
        for (int rr = 0; rr < 8; ++rr)
            sc[rr + 8 * hiL][s0 + n] = c[rr] * 0.125f;
    }
    __syncthreads();

    // Phase 2: softmax -> blend with A_soft -> second softmax (4 rows per wave),
    // final normalized probabilities written as bf16 into sp.
    for (int j = 0; j < 4; ++j) {
        int m = wv * 4 + j;
        float* row = sc[m];
        float mx = -3.0e38f;
        for (int i = lane; i < TT; i += 32) mx = fmaxf(mx, row[i]);
        mx = wred_max(mx);
        float s1 = 0.0f;
        for (int i = lane; i < TT; i += 32) s1 += __expf(row[i] - mx);
        s1 = wred_sum(s1);
        float inv1 = 1.0f / s1;
        const float* arow = As + ((size_t)b * TT + t0 + m) * TT;
        float mx2 = -3.0e38f;
        for (int i = lane; i < TT; i += 32) {
            float bl = 0.5f * (__expf(row[i] - mx) * inv1) + 0.5f * arow[i];
            row[i] = bl;
            mx2 = fmaxf(mx2, bl);
        }
        mx2 = wred_max(mx2);
        float s2 = 0.0f;
        for (int i = lane; i < TT; i += 32) s2 += __expf(row[i] - mx2);
        s2 = wred_sum(s2);
        float inv2 = 1.0f / s2;
        for (int i = lane; i < TT; i += 32)
            sp[m][i] = f2bf(__expf(row[i] - mx2) * inv2);
    }
    __syncthreads();

    // Phase 3: heads = attn @ V (wave wv does d-tile wv); A from LDS bf16.
    int d0 = wv * 16;
    v8f acc = ZERO8;
    size_t voff = (((size_t)h * BB + b) * DD + d0) * TT;
#pragma unroll 8
    for (int kc = 0; kc < 32; ++kc) {
        ABf16 a = load_tile16(&sp[0][0] + kc * 32, TT);
        ABf16 bv = load_tileB16(Vt + voff + kc * 32, TT);
        acc = wmma_bf16(a, bv, acc);
    }
    unsigned short* dst = cat + ((size_t)b * TT + t0) * (HH * DD) + h * DD + d0;
#pragma unroll
    for (int rr = 0; rr < 8; ++rr)
        dst[(size_t)(rr + 8 * hiL) * (HH * DD) + n] = f2bf(acc[rr]);
}

// ---------------- output projection + LN + FFN + LN, fused ----------------
// grid (B*T/16), block 128 (4 waves)
__global__ void k_out(const unsigned short* __restrict__ cat,
                      const unsigned short* __restrict__ WOt, const float* __restrict__ bO,
                      const float* __restrict__ X,
                      const float* __restrict__ g1, const float* __restrict__ be1,
                      const unsigned short* __restrict__ W1t, const float* __restrict__ b1,
                      const unsigned short* __restrict__ W2t, const float* __restrict__ b2,
                      const float* __restrict__ g2, const float* __restrict__ be2,
                      float* __restrict__ out) {
    __shared__ __align__(16) float s_o[16][68];
    __shared__ __align__(16) float s_ln[16][68];
    __shared__ __align__(16) unsigned short s_bf[16][64];
    __shared__ __align__(16) unsigned short s_h1[16][256];
    __shared__ __align__(16) float s_ff[16][68];
    int mt = blockIdx.x;
    int tg0 = mt * 16;
    int wv = __builtin_amdgcn_readfirstlane(threadIdx.x >> 5);
    int lane = threadIdx.x & 31;
    int n = lane & 15, hiL = lane >> 4;

    // GEMM1: out = cat @ WO + bO  (wave wv -> f-tile wv)
    {
        int f0 = wv * 16;
        v8f acc = ZERO8;
        for (int kc = 0; kc < 16; ++kc) {
            ABf16 a = load_tile16(cat + (size_t)tg0 * 512 + kc * 32, 512);
            ABf16 w = load_tileB16(WOt + (size_t)f0 * 512 + kc * 32, 512);
            acc = wmma_bf16(a, w, acc);
        }
        float bia = bO[f0 + n];
#pragma unroll
        for (int rr = 0; rr < 8; ++rr) s_o[rr + 8 * hiL][f0 + n] = acc[rr] + bia;
    }
    __syncthreads();

    // LN1 + residual X
    for (int j = 0; j < 4; ++j) {
        int m = wv * 4 + j;
        float x0 = s_o[m][lane], x1 = s_o[m][lane + 32];
        float mu = wred_sum(x0 + x1) * (1.0f / 64.0f);
        float d0 = x0 - mu, d1 = x1 - mu;
        float var = wred_sum(d0 * d0 + d1 * d1) * (1.0f / 64.0f);
        float rs = rsqrtf(var + 1e-5f);
        const float* xr = X + (size_t)(tg0 + m) * FF;
        float y0 = d0 * rs * g1[lane] + be1[lane] + xr[lane];
        float y1 = d1 * rs * g1[lane + 32] + be1[lane + 32] + xr[lane + 32];
        s_ln[m][lane] = y0;
        s_ln[m][lane + 32] = y1;
        s_bf[m][lane] = f2bf(y0);
        s_bf[m][lane + 32] = f2bf(y1);
    }
    __syncthreads();

    // GEMM2: h1 = gelu(out @ W1 + b1)  (4 n-tiles per wave, N=256)
    for (int q = 0; q < 4; ++q) {
        int c0 = (wv * 4 + q) * 16;
        v8f acc = ZERO8;
        for (int kc = 0; kc < 2; ++kc) {
            ABf16 a = load_tile16(&s_bf[0][0] + kc * 32, 64);
            ABf16 w = load_tileB16(W1t + (size_t)c0 * 64 + kc * 32, 64);
            acc = wmma_bf16(a, w, acc);
        }
        float bia = b1[c0 + n];
#pragma unroll
        for (int rr = 0; rr < 8; ++rr) {
            float x = acc[rr] + bia;
            float gel = 0.5f * x * (1.0f + erff(x * 0.70710678f));
            s_h1[rr + 8 * hiL][c0 + n] = f2bf(gel);
        }
    }
    __syncthreads();

    // GEMM3: ff = h1 @ W2 + b2
    {
        int f0 = wv * 16;
        v8f acc = ZERO8;
        for (int kc = 0; kc < 8; ++kc) {
            ABf16 a = load_tile16(&s_h1[0][0] + kc * 32, 256);
            ABf16 w = load_tileB16(W2t + (size_t)f0 * 256 + kc * 32, 256);
            acc = wmma_bf16(a, w, acc);
        }
        float bia = b2[f0 + n];
#pragma unroll
        for (int rr = 0; rr < 8; ++rr) s_ff[rr + 8 * hiL][f0 + n] = acc[rr] + bia;
    }
    __syncthreads();

    // LN2 + residual(out)
    for (int j = 0; j < 4; ++j) {
        int m = wv * 4 + j;
        float x0 = s_ff[m][lane], x1 = s_ff[m][lane + 32];
        float mu = wred_sum(x0 + x1) * (1.0f / 64.0f);
        float d0 = x0 - mu, d1 = x1 - mu;
        float var = wred_sum(d0 * d0 + d1 * d1) * (1.0f / 64.0f);
        float rs = rsqrtf(var + 1e-5f);
        float* orow = out + (size_t)(tg0 + m) * FF;
        orow[lane] = d0 * rs * g2[lane] + be2[lane] + s_ln[m][lane];
        orow[lane + 32] = d1 * rs * g2[lane + 32] + be2[lane + 32] + s_ln[m][lane + 32];
    }
}

// ---------------- host launcher ----------------

extern "C" void kernel_launch(void* const* d_in, const int* in_sizes, int n_in,
                              void* d_out, int out_size, void* d_ws, size_t ws_size,
                              hipStream_t stream) {
    const float* X  = (const float*)d_in[0];
    const float* At = (const float*)d_in[1];
    const float* WQ = (const float*)d_in[2];
    const float* bQ = (const float*)d_in[3];
    const float* WK = (const float*)d_in[4];
    const float* bK = (const float*)d_in[5];
    const float* WV = (const float*)d_in[6];
    const float* bV = (const float*)d_in[7];
    const float* WO = (const float*)d_in[8];
    const float* bO = (const float*)d_in[9];
    const float* g1 = (const float*)d_in[10];
    const float* be1 = (const float*)d_in[11];
    const float* W1 = (const float*)d_in[12];
    const float* b1 = (const float*)d_in[13];
    const float* W2 = (const float*)d_in[14];
    const float* b2 = (const float*)d_in[15];
    const float* g2 = (const float*)d_in[16];
    const float* be2 = (const float*)d_in[17];
    float* out = (float*)d_out;

    char* w = (char*)d_ws;
    unsigned short* Xbf = (unsigned short*)w;  w += (size_t)BB * TT * FF * 2;
    unsigned short* Wqt = (unsigned short*)w;  w += (size_t)HH * 64 * 64 * 2;
    unsigned short* Wkt = (unsigned short*)w;  w += (size_t)HH * 64 * 64 * 2;
    unsigned short* Wvt = (unsigned short*)w;  w += (size_t)HH * 64 * 64 * 2;
    unsigned short* WOt = (unsigned short*)w;  w += (size_t)64 * 512 * 2;
    unsigned short* W1t = (unsigned short*)w;  w += (size_t)256 * 64 * 2;
    unsigned short* W2t = (unsigned short*)w;  w += (size_t)64 * 256 * 2;
    w = (char*)(((size_t)w + 255) & ~(size_t)255);
    unsigned short* Qbf = (unsigned short*)w;  w += (size_t)HH * BB * TT * DD * 2;
    unsigned short* Kbf = (unsigned short*)w;  w += (size_t)HH * BB * TT * DD * 2;
    unsigned short* Vt  = (unsigned short*)w;  w += (size_t)HH * BB * TT * DD * 2;
    unsigned short* cat = (unsigned short*)w;  w += (size_t)BB * TT * HH * DD * 2;
    w = (char*)(((size_t)w + 255) & ~(size_t)255);
    float* As = (float*)w;                     w += (size_t)BB * TT * TT * 4;

    // conversions
    k_cvt<<<(BB * TT * FF + 255) / 256, 256, 0, stream>>>(X, Xbf, BB * TT * FF);
    k_cvt_t<<<dim3(16, HH), 256, 0, stream>>>(WQ, Wqt, 64, 64);
    k_cvt_t<<<dim3(16, HH), 256, 0, stream>>>(WK, Wkt, 64, 64);
    k_cvt_t<<<dim3(16, HH), 256, 0, stream>>>(WV, Wvt, 64, 64);
    k_cvt_t<<<dim3(128, 1), 256, 0, stream>>>(WO, WOt, 512, 64);
    k_cvt_t<<<dim3(64, 1), 256, 0, stream>>>(W1, W1t, 64, 256);
    k_cvt_t<<<dim3(64, 1), 256, 0, stream>>>(W2, W2t, 256, 64);

    // head-independent softmax(relu(A_temp))
    k_asoft<<<BB * TT, 256, 0, stream>>>(At, As);

    // QKV projections
    k_qkv<<<dim3(BB * TT / 16, HH), 96, 0, stream>>>(Xbf, Wqt, Wkt, Wvt,
                                                     bQ, bK, bV, Qbf, Kbf, Vt);

    // attention
    k_attn<<<dim3(TT / 16, HH * BB), 128, 0, stream>>>(Qbf, Kbf, Vt, As, cat);

    // output projection + LN + FFN + LN
    k_out<<<BB * TT / 16, 128, 0, stream>>>(cat, WOt, bO, X, g1, be1,
                                            W1t, b1, W2t, b2, g2, be2, out);
}
